// Rotation_9242769622431
// MI455X (gfx1250) — matrix operand order repaired and checked by reading, not compiled, not run on hardware
//
#include <hip/hip_runtime.h>

// Problem constants (from reference)
#define DIM    4096
#define GROUP  128
#define KROT   8
#define NGRP   (DIM / GROUP)      // 32 groups
#define HPL    (DIM / 2)          // theta per layer = 2048

typedef __attribute__((ext_vector_type(16))) _Float16 v16h;
typedef __attribute__((ext_vector_type(8)))  _Float16 v8h;
typedef __attribute__((ext_vector_type(8)))  float    v8f;

// ---------------------------------------------------------------------------
// Kernel 1: build per-group composite rotation matrix M_g = diag(scale)*R_g
// and emit it pre-swizzled into WMMA B-fragment layout (32x16 f16 frags).
//
// ws layout (halves): bmat[g][kt(4)][nt(8)][lane(32)][h(16)]
//   lane L of frag(kt,nt) holds B[kk][nn] = M[nt*16+nn][kt*32+kk]
//   with nn = L%16, kk = (h/8)*16 + (L/16)*8 + h%8   (same k order as A frags)
// ---------------------------------------------------------------------------
__global__ __launch_bounds__(GROUP) void build_rot_kernel(
    const int* __restrict__ pairs,          // (KROT, DIM) int32
    const _Float16* __restrict__ theta,     // (KROT, DIM/2) f16
    const _Float16* __restrict__ scales,    // (1, DIM) f16
    _Float16* __restrict__ bmat)            // ws: NGRP*GROUP*GROUP halves
{
    __shared__ float R[GROUP][GROUP];       // 64 KB, [row][col]
    const int g = blockIdx.x;
    const int t = threadIdx.x;              // this thread owns column t

    // identity
    for (int j = 0; j < GROUP; ++j) R[j][t] = (j == t) ? 1.0f : 0.0f;
    // Rotation layers: pairs within a layer are disjoint (a permutation) and
    // each thread touches only its own column -> no barriers needed here.
    for (int k = 0; k < KROT; ++k) {
        const int*      pk  = pairs + k * DIM + g * GROUP;     // values in [0,128)
        const _Float16* thk = theta + k * HPL + g * (GROUP / 2);
        for (int i = 0; i < GROUP / 2; ++i) {
            const int a = pk[2 * i];
            const int b = pk[2 * i + 1];
            const float th = (float)thk[i];
            const float c = cosf(th), s = sinf(th);
            const float ra = R[a][t], rb = R[b][t];
            R[a][t] = c * ra - s * rb;      // G_k applied: R <- G_k * R
            R[b][t] = s * ra + c * rb;
        }
    }
    __syncthreads();                        // emit phase reads other columns

    // Emit scaled matrix in B-fragment order.
    _Float16* out = bmat + (size_t)g * GROUP * GROUP;
    for (int q = t; q < GROUP * GROUP; q += GROUP) {
        const int h  = q & 15;
        const int L  = (q >> 4) & 31;
        const int nt = (q >> 9) & 7;
        const int kt = (q >> 12) & 3;
        const int j  = nt * 16 + (L & 15);              // output channel (row of M)
        const int i  = kt * 32 + (h >> 3) * 16 + (L >> 4) * 8 + (h & 7);     // input channel (col of M)
        out[q] = (_Float16)(R[j][i] * (float)scales[g * GROUP + j]);
    }
}

// ---------------------------------------------------------------------------
// Kernel 2: block-diagonal GEMM via v_wmma_f32_16x16x32_f16.
// One wave handles 32 rows x one 128-channel group:
//   2 row-tiles x 8 n-tiles x 4 k-tiles = 64 WMMAs per wave.
// Results transposed through per-wave LDS staging for coalesced b128 stores.
// ---------------------------------------------------------------------------
__global__ __launch_bounds__(256) void rot_gemm_kernel(
    const _Float16* __restrict__ x,         // (N, DIM) f16
    const _Float16* __restrict__ bmat,      // swizzled matrices (ws)
    _Float16* __restrict__ out)             // (N, DIM) f16
{
    __shared__ __align__(16) _Float16 stage[8][32 * GROUP];   // 8 waves * 8KB = 64KB

    const int lane    = threadIdx.x & 31;
    const int wave    = threadIdx.x >> 5;
    const int g       = blockIdx.y;                       // group 0..31
    const int rowbase = (blockIdx.x * 8 + wave) * 32;     // 32 rows per wave
    const int lmod    = lane & 15;
    const int halfsel = lane >> 4;                        // 0: lanes 0-15, 1: 16-31

    // --- Load A fragments (ISA 16x32 f16 A layout):
    // lane<16  (M=lane):    VGPR0-3 = K0..7,  VGPR4-7 = K16..23
    // lane>=16 (M=lane-16): VGPR0-3 = K8..15, VGPR4-7 = K24..31
    v16h afrag[2][4];
#pragma unroll
    for (int rt = 0; rt < 2; ++rt) {
#pragma unroll
        for (int kt = 0; kt < 4; ++kt) {
            const _Float16* base = x + (size_t)(rowbase + rt * 16 + lmod) * DIM
                                     + g * GROUP + kt * 32 + halfsel * 8;
            union { v16h v; v8h h[2]; } u;
            u.h[0] = *(const v8h*)(base);        // frag halves 0..7  (16B)
            u.h[1] = *(const v8h*)(base + 16);   // frag halves 8..15 (16B)
            afrag[rt][kt] = u.v;
        }
    }

    const _Float16* bg = bmat + (size_t)g * GROUP * GROUP;   // resident in L2

#pragma unroll
    for (int nt = 0; nt < 8; ++nt) {
        v8f acc0 = {};
        v8f acc1 = {};
#pragma unroll
        for (int kt = 0; kt < 4; ++kt) {
            // B fragment: pre-swizzled, 32B contiguous per lane
            const _Float16* bb = bg + (((kt * 8 + nt) * 32 + lane) * 16);
            union { v16h v; v8h h[2]; } u;
            u.h[0] = *(const v8h*)(bb);
            u.h[1] = *(const v8h*)(bb + 8);
            const v16h bfrag = u.v;
            acc0 = __builtin_amdgcn_wmma_f32_16x16x32_f16(
                       false, afrag[0][kt], false, bfrag, (short)0, acc0, false, false);
            acc1 = __builtin_amdgcn_wmma_f32_16x16x32_f16(
                       false, afrag[1][kt], false, bfrag, (short)0, acc1, false, false);
        }
        // D layout: VGPR r, lanes 0-15 -> (M=r, N=lane); lanes 16-31 -> (M=r+8, N=lane-16)
        _Float16* st = &stage[wave][0];
#pragma unroll
        for (int r = 0; r < 8; ++r) {
            const int m = r + halfsel * 8;
            st[(m)      * GROUP + nt * 16 + lmod] = (_Float16)acc0[r];
            st[(16 + m) * GROUP + nt * 16 + lmod] = (_Float16)acc1[r];
        }
    }

    // Per-wave staging region -> only need this wave's DS ops complete.
    asm volatile("s_wait_dscnt 0" ::: "memory");

    // Coalesced write-out: 32 rows x 256B = 512 x 16B chunks, 16 iters x 32 lanes.
#pragma unroll
    for (int it = 0; it < 16; ++it) {
        const int c  = it * 32 + lane;
        const int rl = c >> 4;        // local row 0..31
        const int cc = c & 15;        // 16B chunk within row
        const int4 v = *(const int4*)(&stage[wave][rl * GROUP + cc * 8]);
        *(int4*)(out + (size_t)(rowbase + rl) * DIM + g * GROUP + cc * 8) = v;
    }
}

// ---------------------------------------------------------------------------
extern "C" void kernel_launch(void* const* d_in, const int* in_sizes, int n_in,
                              void* d_out, int out_size, void* d_ws, size_t ws_size,
                              hipStream_t stream) {
    const _Float16* x      = (const _Float16*)d_in[0];   // (N, DIM) f16
    const int*      pairs  = (const int*)d_in[1];        // (KROT, DIM) i32
    const _Float16* theta  = (const _Float16*)d_in[2];   // (KROT, DIM/2) f16
    const _Float16* scales = (const _Float16*)d_in[3];   // (1, DIM) f16
    _Float16*       out    = (_Float16*)d_out;
    _Float16*       bmat   = (_Float16*)d_ws;            // 1 MB: 32 * 128*128 f16

    const int N = in_sizes[0] / DIM;                     // 16384

    // 1) Build the 32 composite 128x128 rotation matrices (scale folded in).
    build_rot_kernel<<<dim3(NGRP), dim3(GROUP), 0, stream>>>(pairs, theta, scales, bmat);

    // 2) Block-diagonal WMMA GEMM: grid (N/256 row-blocks, 32 groups), 8 waves/block.
    rot_gemm_kernel<<<dim3(N / 256, NGRP), dim3(256), 0, stream>>>(x, bmat, out);
}
